// SerriformBlock_13795434954813
// MI455X (gfx1250) — matrix operand order repaired and checked by reading, not compiled, not run on hardware
//
#include <hip/hip_runtime.h>
#include <hip/hip_bf16.h>

typedef __attribute__((ext_vector_type(16))) __bf16 v16bf;
typedef __attribute__((ext_vector_type(8)))  float  v8f;
typedef unsigned short u16;

static constexpr int MD = 1024;   // model dim
static constexpr int NE = 8;      // experts
static constexpr int NT = 8192;   // B*S tokens
static constexpr int NS = 2048;   // seq len
static constexpr int KSTR = 40;   // LDS row stride in bf16 units (32 + 8 pad -> 80B, 16B aligned, conflict-free)

__device__ __forceinline__ float siluf(float v) { return v / (1.0f + __expf(-v)); }

// ---- CDNA5 async global->LDS copy (ASYNCcnt-tracked, no VGPR round trip) ----
__device__ __forceinline__ void async_ld16(unsigned ldsOff, const void* g) {
    unsigned long long ga = (unsigned long long)(uintptr_t)g;
    asm volatile("global_load_async_to_lds_b128 %0, %1, off"
                 :: "v"(ldsOff), "v"(ga) : "memory");
}
__device__ __forceinline__ void wait_async0() {
    asm volatile("s_wait_asynccnt 0x0" ::: "memory");
}
__device__ __forceinline__ unsigned lds_off(const void* p) {
    // flat shared-aperture address keeps the LDS offset in addr[31:0]
    return (unsigned)(uintptr_t)p;
}

// ---------------- elementwise helpers ----------------

__global__ void k_f2bf(const float* __restrict__ in, __hip_bfloat16* __restrict__ out, int n) {
    int i = blockIdx.x * blockDim.x + threadIdx.x;
    if (i < n) out[i] = __float2bfloat16(in[i]);
}

__global__ void k_zero16(int* __restrict__ p) {
    if (threadIdx.x < 16) p[threadIdx.x] = 0;
}

// generic rmsnorm: y = w * x / rms(x)
__global__ void k_rmsnorm(const float* __restrict__ x, const float* __restrict__ w,
                          float* __restrict__ y) {
    int t = blockIdx.x;
    const float* xr = x + (size_t)t * MD;
    float* yr = y + (size_t)t * MD;
    __shared__ float red[256];
    float s = 0.f;
    for (int d = threadIdx.x; d < MD; d += 256) { float v = xr[d]; s += v * v; }
    red[threadIdx.x] = s;
    __syncthreads();
    for (int st = 128; st > 0; st >>= 1) {
        if (threadIdx.x < st) red[threadIdx.x] += red[threadIdx.x + st];
        __syncthreads();
    }
    float rinv = rsqrtf(red[0] / (float)MD + 1e-6f);
    for (int d = threadIdx.x; d < MD; d += 256) yr[d] = w[d] * xr[d] * rinv;
}

// causal depthwise conv K=3 (torch pad-left-2 semantics), outputs bf16 GEMM-A operand
__global__ void k_dwconv(const float* __restrict__ h1, const float* __restrict__ dww,
                         const float* __restrict__ dwb, __hip_bfloat16* __restrict__ c) {
    size_t i = (size_t)blockIdx.x * blockDim.x + threadIdx.x;
    if (i >= (size_t)NT * MD) return;
    int d = (int)(i % MD);
    size_t t = i / MD;
    int l = (int)(t % NS);
    float acc = dwb[d];
    float w0 = dww[d * 3 + 0], w1 = dww[d * 3 + 1], w2 = dww[d * 3 + 2];
    acc += w2 * h1[i];
    if (l >= 1) acc += w1 * h1[i - MD];
    if (l >= 2) acc += w0 * h1[i - 2 * MD];
    c[i] = __float2bfloat16(acc);
}

// ---------------- WMMA fragment loaders (wave32, 16x16x32 bf16) ----------------
// A-matrix 16x32: lane = h*16 + m (m=lane&15); elems 0..7 -> K = 8h+[0..7]; 8..15 -> K = 16+8h+[0..7]
__device__ __forceinline__ v16bf fragA(const u16* tile, int rowBase, int lane) {
    int r = lane & 15, h = lane >> 4;
    const u16* p = tile + (rowBase + r) * KSTR + h * 8;
    union { uint4 q[2]; v16bf v; } u;
    u.q[0] = *(const uint4*)p;
    u.q[1] = *(const uint4*)(p + 16);
    return u.v;
}
// B-matrix 32x16: lane = h*16 + n; elems 0..7 -> K = 8h+[0..7]; 8..15 -> K = 16+8h+[0..7]
__device__ __forceinline__ v16bf fragB(const u16* tile, int rowBase, int lane) {
    int n = lane & 15, h = lane >> 4;
    const u16* p = tile + (rowBase + n) * KSTR + h * 8;
    union { uint4 q[2]; v16bf v; } u;
    u.q[0] = *(const uint4*)p;
    u.q[1] = *(const uint4*)(p + 16);
    return u.v;
}

#define WMMA_BF16(A_, B_, C_) \
    __builtin_amdgcn_wmma_f32_16x16x32_bf16(false, (A_), false, (B_), (short)0, (C_), false, false)

// ---------------- GEMM 1: y = silu(conv @ pw_w^T + pw_b); h = x + y ----------------
__global__ void __launch_bounds__(128)
k_gemm_pw(const __hip_bfloat16* __restrict__ A, const __hip_bfloat16* __restrict__ W,
          const float* __restrict__ bias, const float* __restrict__ x,
          float* __restrict__ h, __hip_bfloat16* __restrict__ hbf) {
    __shared__ u16 As[2][64 * KSTR];
    __shared__ u16 Bs[2][64 * KSTR];
    const int tid = threadIdx.x, lane = tid & 31, wave = tid >> 5;
    const int wm = wave >> 1, wn = wave & 1;
    const int rowBase = blockIdx.x * 64, colBase = blockIdx.y * 64;

    auto issue = [&](int k0, int buf) {
#pragma unroll
        for (int it = 0; it < 2; ++it) {
            int u = tid + it * 128;
            int r = u >> 2, q = u & 3;
            async_ld16(lds_off(&As[buf][r * KSTR + q * 8]),
                       A + (size_t)(rowBase + r) * MD + k0 + q * 8);
            async_ld16(lds_off(&Bs[buf][r * KSTR + q * 8]),
                       W + (size_t)(colBase + r) * MD + k0 + q * 8);
        }
    };

    v8f acc[2][2] = {};
    issue(0, 0);
    for (int k0 = 0; k0 < MD; k0 += 32) {
        int cur = (k0 >> 5) & 1;
        wait_async0();
        __syncthreads();
        if (k0 + 32 < MD) issue(k0 + 32, cur ^ 1);
        v16bf a0 = fragA(As[cur], wm * 32, lane);
        v16bf a1 = fragA(As[cur], wm * 32 + 16, lane);
        v16bf b0 = fragB(Bs[cur], wn * 32, lane);
        v16bf b1 = fragB(Bs[cur], wn * 32 + 16, lane);
        acc[0][0] = WMMA_BF16(a0, b0, acc[0][0]);
        acc[0][1] = WMMA_BF16(a0, b1, acc[0][1]);
        acc[1][0] = WMMA_BF16(a1, b0, acc[1][0]);
        acc[1][1] = WMMA_BF16(a1, b1, acc[1][1]);
    }
    int n = lane & 15, mh = lane >> 4;
    for (int mi = 0; mi < 2; mi++)
        for (int ni = 0; ni < 2; ni++)
            for (int r = 0; r < 8; r++) {
                int m = r + 8 * mh;
                int row = rowBase + wm * 32 + mi * 16 + m;
                int col = colBase + wn * 32 + ni * 16 + n;
                float v = acc[mi][ni][r] + bias[col];
                float hh = x[(size_t)row * MD + col] + siluf(v);
                h[(size_t)row * MD + col] = hh;
                hbf[(size_t)row * MD + col] = __float2bfloat16(hh);
            }
}

// ---------------- router: wave per token, top-2 softmax ----------------
__global__ void __launch_bounds__(128)
k_router(const float* __restrict__ h, const float* __restrict__ rw, const float* __restrict__ rb,
         int* __restrict__ counts, int2* __restrict__ tokE, float2* __restrict__ tokW) {
    int t = blockIdx.x * 4 + (threadIdx.x >> 5);
    int lane = threadIdx.x & 31;
    const float4* hr = (const float4*)(h + (size_t)t * MD);
    float dot[NE] = {};
    for (int i = lane; i < MD / 4; i += 32) {
        float4 hv = hr[i];
#pragma unroll
        for (int e = 0; e < NE; e++) {
            float4 wv = ((const float4*)(rw + e * MD))[i];
            dot[e] += hv.x * wv.x + hv.y * wv.y + hv.z * wv.z + hv.w * wv.w;
        }
    }
#pragma unroll
    for (int e = 0; e < NE; e++)
        for (int off = 16; off > 0; off >>= 1) dot[e] += __shfl_xor(dot[e], off, 32);
    if (lane == 0) {
        float lg[NE];
        for (int e = 0; e < NE; e++) lg[e] = dot[e] + rb[e];
        int e0 = 0;
        for (int e = 1; e < NE; e++) if (lg[e] > lg[e0]) e0 = e;
        int e1 = (e0 == 0) ? 1 : 0;
        for (int e = 0; e < NE; e++) if (e != e0 && lg[e] > lg[e1]) e1 = e;
        float x0 = 1.0f, x1 = __expf(lg[e1] - lg[e0]);
        float inv = 1.0f / (x0 + x1);
        atomicAdd(&counts[e0], 1);
        atomicAdd(&counts[e1], 1);
        tokE[t] = make_int2(e0, e1);
        tokW[t] = make_float2(x0 * inv, x1 * inv);
    }
}

__global__ void k_offsets(const int* __restrict__ counts, int* __restrict__ offsets) {
    if (threadIdx.x == 0) {
        int s = 0;
        for (int e = 0; e < NE; e++) { offsets[e] = s; s += counts[e]; }
        offsets[NE] = s;
    }
}

__global__ void k_scatter(const int2* __restrict__ tokE, const float2* __restrict__ tokW,
                          const int* __restrict__ offsets, int* __restrict__ cursors,
                          int* __restrict__ list, float* __restrict__ gateS,
                          int* __restrict__ tokslot) {
    int t = blockIdx.x * blockDim.x + threadIdx.x;
    if (t >= NT) return;
    int2 e = tokE[t];
    float2 w = tokW[t];
    int s0 = offsets[e.x] + atomicAdd(&cursors[e.x], 1);
    list[s0] = t; gateS[s0] = w.x; tokslot[2 * t] = s0;
    int s1 = offsets[e.y] + atomicAdd(&cursors[e.y], 1);
    list[s1] = t; gateS[s1] = w.y; tokslot[2 * t + 1] = s1;
}

// ---------------- per-expert gather GEMM: slotOut[slot] = gate * silu(h @ exp_w[e]^T + exp_b[e]) ----
__global__ void __launch_bounds__(128)
k_gemm_exp(const __hip_bfloat16* __restrict__ hbf, const __hip_bfloat16* __restrict__ Wexp,
           const float* __restrict__ expb, const int* __restrict__ offsets,
           const int* __restrict__ counts, const int* __restrict__ list,
           const float* __restrict__ gateS, float* __restrict__ slotOut) {
    int e = blockIdx.z;
    int cnt = counts[e];
    int tileRow = blockIdx.x * 64;
    if (tileRow >= cnt) return;
    int slotBase = offsets[e] + tileRow;
    int colBase = blockIdx.y * 64;

    __shared__ u16 As[2][64 * KSTR];
    __shared__ u16 Bs[2][64 * KSTR];
    __shared__ int toks[64];
    __shared__ float gts[64];

    const int tid = threadIdx.x, lane = tid & 31, wave = tid >> 5;
    const int wm = wave >> 1, wn = wave & 1;
    if (tid < 64) {
        int i = tileRow + tid;
        bool ok = i < cnt;
        toks[tid] = ok ? list[slotBase + tid] : 0;   // invalid rows fetch token 0 (finite junk)
        gts[tid] = ok ? gateS[slotBase + tid] : 0.f;
    }
    __syncthreads();

    const __hip_bfloat16* We = Wexp + (size_t)e * MD * MD;
    auto issue = [&](int k0, int buf) {
#pragma unroll
        for (int it = 0; it < 2; ++it) {
            int u = tid + it * 128;
            int r = u >> 2, q = u & 3;
            async_ld16(lds_off(&As[buf][r * KSTR + q * 8]),
                       hbf + (size_t)toks[r] * MD + k0 + q * 8);
            async_ld16(lds_off(&Bs[buf][r * KSTR + q * 8]),
                       We + (size_t)(colBase + r) * MD + k0 + q * 8);
        }
    };

    v8f acc[2][2] = {};
    issue(0, 0);
    for (int k0 = 0; k0 < MD; k0 += 32) {
        int cur = (k0 >> 5) & 1;
        wait_async0();
        __syncthreads();
        if (k0 + 32 < MD) issue(k0 + 32, cur ^ 1);
        v16bf a0 = fragA(As[cur], wm * 32, lane);
        v16bf a1 = fragA(As[cur], wm * 32 + 16, lane);
        v16bf b0 = fragB(Bs[cur], wn * 32, lane);
        v16bf b1 = fragB(Bs[cur], wn * 32 + 16, lane);
        acc[0][0] = WMMA_BF16(a0, b0, acc[0][0]);
        acc[0][1] = WMMA_BF16(a0, b1, acc[0][1]);
        acc[1][0] = WMMA_BF16(a1, b0, acc[1][0]);
        acc[1][1] = WMMA_BF16(a1, b1, acc[1][1]);
    }
    int n = lane & 15, mh = lane >> 4;
    for (int mi = 0; mi < 2; mi++)
        for (int ni = 0; ni < 2; ni++)
            for (int r = 0; r < 8; r++) {
                int m = r + 8 * mh;
                int rowIdx = wm * 32 + mi * 16 + m;
                if (tileRow + rowIdx < cnt) {
                    int col = colBase + wn * 32 + ni * 16 + n;
                    float v = acc[mi][ni][r] + expb[e * MD + col];
                    slotOut[(size_t)(slotBase + rowIdx) * MD + col] = siluf(v) * gts[rowIdx];
                }
            }
}

// combine two slots per token -> bf16 A operand for output projection
__global__ void k_combine(const float* __restrict__ slotOut, const int* __restrict__ tokslot,
                          __hip_bfloat16* __restrict__ A2) {
    size_t i = (size_t)blockIdx.x * blockDim.x + threadIdx.x;
    if (i >= (size_t)NT * MD) return;
    int d = (int)(i % MD);
    size_t t = i / MD;
    int s0 = tokslot[2 * t], s1 = tokslot[2 * t + 1];
    A2[i] = __float2bfloat16(slotOut[(size_t)s0 * MD + d] + slotOut[(size_t)s1 * MD + d]);
}

// ---------------- output projection GEMM: pre = combined @ out_w^T + out_b + h ----------------
__global__ void __launch_bounds__(128)
k_gemm_out(const __hip_bfloat16* __restrict__ A, const __hip_bfloat16* __restrict__ W,
           const float* __restrict__ bias, const float* __restrict__ h,
           float* __restrict__ pre) {
    __shared__ u16 As[2][64 * KSTR];
    __shared__ u16 Bs[2][64 * KSTR];
    const int tid = threadIdx.x, lane = tid & 31, wave = tid >> 5;
    const int wm = wave >> 1, wn = wave & 1;
    const int rowBase = blockIdx.x * 64, colBase = blockIdx.y * 64;

    auto issue = [&](int k0, int buf) {
#pragma unroll
        for (int it = 0; it < 2; ++it) {
            int u = tid + it * 128;
            int r = u >> 2, q = u & 3;
            async_ld16(lds_off(&As[buf][r * KSTR + q * 8]),
                       A + (size_t)(rowBase + r) * MD + k0 + q * 8);
            async_ld16(lds_off(&Bs[buf][r * KSTR + q * 8]),
                       W + (size_t)(colBase + r) * MD + k0 + q * 8);
        }
    };

    v8f acc[2][2] = {};
    issue(0, 0);
    for (int k0 = 0; k0 < MD; k0 += 32) {
        int cur = (k0 >> 5) & 1;
        wait_async0();
        __syncthreads();
        if (k0 + 32 < MD) issue(k0 + 32, cur ^ 1);
        v16bf a0 = fragA(As[cur], wm * 32, lane);
        v16bf a1 = fragA(As[cur], wm * 32 + 16, lane);
        v16bf b0 = fragB(Bs[cur], wn * 32, lane);
        v16bf b1 = fragB(Bs[cur], wn * 32 + 16, lane);
        acc[0][0] = WMMA_BF16(a0, b0, acc[0][0]);
        acc[0][1] = WMMA_BF16(a0, b1, acc[0][1]);
        acc[1][0] = WMMA_BF16(a1, b0, acc[1][0]);
        acc[1][1] = WMMA_BF16(a1, b1, acc[1][1]);
    }
    int n = lane & 15, mh = lane >> 4;
    for (int mi = 0; mi < 2; mi++)
        for (int ni = 0; ni < 2; ni++)
            for (int r = 0; r < 8; r++) {
                int m = r + 8 * mh;
                int row = rowBase + wm * 32 + mi * 16 + m;
                int col = colBase + wn * 32 + ni * 16 + n;
                pre[(size_t)row * MD + col] = acc[mi][ni][r] + bias[col] + h[(size_t)row * MD + col];
            }
}

// ---------------- host side ----------------
extern "C" void kernel_launch(void* const* d_in, const int* in_sizes, int n_in,
                              void* d_out, int out_size, void* d_ws, size_t ws_size,
                              hipStream_t stream) {
    const float* x   = (const float*)d_in[0];
    const float* n1w = (const float*)d_in[1];
    const float* dww = (const float*)d_in[2];
    const float* dwb = (const float*)d_in[3];
    const float* pww = (const float*)d_in[4];
    const float* pwb = (const float*)d_in[5];
    const float* rw  = (const float*)d_in[6];
    const float* rb  = (const float*)d_in[7];
    const float* ew  = (const float*)d_in[8];
    const float* eb  = (const float*)d_in[9];
    const float* ow  = (const float*)d_in[10];
    const float* ob  = (const float*)d_in[11];
    const float* n2w = (const float*)d_in[12];
    float* out = (float*)d_out;

    char* p = (char*)d_ws;
    auto take = [&](size_t nbytes) { char* r = p; p += (nbytes + 255) & ~(size_t)255; return r; };

    __hip_bfloat16* pwWbf  = (__hip_bfloat16*)take((size_t)MD * MD * 2);
    __hip_bfloat16* outWbf = (__hip_bfloat16*)take((size_t)MD * MD * 2);
    __hip_bfloat16* expWbf = (__hip_bfloat16*)take((size_t)NE * MD * MD * 2);
    float*          h1     = (float*)take((size_t)NT * MD * 4);   // reused as `pre` after dwconv
    __hip_bfloat16* actbf  = (__hip_bfloat16*)take((size_t)NT * MD * 2); // A0 then A2
    float*          h      = (float*)take((size_t)NT * MD * 4);
    __hip_bfloat16* hbf    = (__hip_bfloat16*)take((size_t)NT * MD * 2);
    float*          slotO  = (float*)take((size_t)2 * NT * MD * 4);
    int2*           tokE   = (int2*)take((size_t)NT * 8);
    float2*         tokW   = (float2*)take((size_t)NT * 8);
    int*            list   = (int*)take((size_t)2 * NT * 4);
    float*          gateS  = (float*)take((size_t)2 * NT * 4);
    int*            tokslot= (int*)take((size_t)2 * NT * 4);
    int*            meta   = (int*)take(256);
    int* counts  = meta;        // [8]
    int* cursors = meta + 8;    // [8]
    int* offsets = meta + 16;   // [9]
    float* pre = h1;

    const size_t nTD = (size_t)NT * MD;

    // weights -> bf16
    k_f2bf<<<(MD * MD + 255) / 256, 256, 0, stream>>>(pww, pwWbf, MD * MD);
    k_f2bf<<<(MD * MD + 255) / 256, 256, 0, stream>>>(ow, outWbf, MD * MD);
    k_f2bf<<<(NE * MD * MD + 255) / 256, 256, 0, stream>>>(ew, expWbf, NE * MD * MD);
    k_zero16<<<1, 32, 0, stream>>>(meta);

    // conv branch
    k_rmsnorm<<<NT, 256, 0, stream>>>(x, n1w, h1);
    k_dwconv<<<(int)((nTD + 255) / 256), 256, 0, stream>>>(h1, dww, dwb, actbf);
    k_gemm_pw<<<dim3(NT / 64, MD / 64), 128, 0, stream>>>(actbf, pwWbf, pwb, x, h, hbf);

    // MoE routing
    k_router<<<NT / 4, 128, 0, stream>>>(h, rw, rb, counts, tokE, tokW);
    k_offsets<<<1, 32, 0, stream>>>(counts, offsets);
    k_scatter<<<(NT + 255) / 256, 256, 0, stream>>>(tokE, tokW, offsets, cursors, list, gateS, tokslot);

    // expert GEMMs (sparse: only routed tokens; empty tiles exit early)
    k_gemm_exp<<<dim3(NT / 64, MD / 64, NE), 128, 0, stream>>>(hbf, expWbf, eb, offsets, counts,
                                                               list, gateS, slotO);
    k_combine<<<(int)((nTD + 255) / 256), 256, 0, stream>>>(slotO, tokslot, actbf);

    // output projection + residual + final rmsnorm
    k_gemm_out<<<dim3(NT / 64, MD / 64), 128, 0, stream>>>(actbf, outWbf, ob, h, pre);
    k_rmsnorm<<<NT, 256, 0, stream>>>(pre, n2w, out);
}
